// TabNet_18502719111485
// MI455X (gfx1250) — matrix operand order repaired and checked by reading, not compiled, not run on hardware
//
#include <hip/hip_runtime.h>
#include <hip/hip_bf16.h>

// ---------------------------------------------------------------------------
// TabNet forward on gfx1250 (MI455X).
// - bf16 activations between layers (halves HBM traffic)
// - weights pre-packed into WMMA B-fragment lane layout -> B frags are two
//   contiguous global_load_b128 straight from L2 (no LDS for B)
// - A tiles double-buffered in LDS via GLOBAL_LOAD_ASYNC_TO_LDS_B128:
//   next tile's async copy overlaps current tile's WMMAs (ASYNCcnt discipline)
// - 128M block tiles: each wave = 32M x 64N -> every B fragment feeds 2 WMMAs
// - fused epilogues: relu/BN/GLU/residual, attention prior product
// ---------------------------------------------------------------------------

typedef __bf16 bf16;
typedef __attribute__((ext_vector_type(16))) __bf16 v16bf;
typedef __attribute__((ext_vector_type(8)))  __bf16 v8bf;
typedef __attribute__((ext_vector_type(8)))  float  v8f;

#define BATCH   32768L
#define FDIM    512
#define ADIM    256
#define ODIM    128
#define NSTEPS  5
#define SQRT_H  0.70710678118654752f

static constexpr int LDS_A_STRIDE = 40;  // bf16 elems per A row (32 used) -> 80B, 16B aligned

// Async 16B global -> LDS copy (ASYNCcnt tracked).
// LDS byte offset = low 32 bits of the generic pointer (LDS aperture rule).
__device__ __forceinline__ void async_copy16(void* lds_dst, const void* gsrc) {
    const unsigned loff = (unsigned)(unsigned long long)lds_dst;
    asm volatile("global_load_async_to_lds_b128 %0, %1, off"
                 :: "v"(loff), "v"(gsrc) : "memory");
}
template <int N>
__device__ __forceinline__ void async_waitN() {
    asm volatile("s_wait_asynccnt %0" :: "i"(N) : "memory");
}

// A fragment: 16x32 bf16 (MxK).  lanes 0-15: row=lane, K 0..7 & 16..23
//                                lanes 16-31: row=lane-16, K 8..15 & 24..31
__device__ __forceinline__ v16bf load_a_frag(const bf16* __restrict__ Atile, int lane) {
    const int row = lane & 15;
    const int k0  = (lane < 16) ? 0 : 8;
    const bf16* p = Atile + row * LDS_A_STRIDE + k0;
    v8bf lo = *(const v8bf*)(p);
    v8bf hi = *(const v8bf*)(p + 16);
    v16bf r;
#pragma unroll
    for (int j = 0; j < 8; ++j) { r[j] = lo[j]; r[j + 8] = hi[j]; }
    return r;
}

// B fragment from packed weights: tile = 32 lanes x 16 bf16 contiguous (1KB).
// lane l holds K = (l<16?0:8)+e+(e&8), N = l&15  (per ISA 16-bit B map).
__device__ __forceinline__ v16bf load_b_frag_g(const bf16* __restrict__ tile, int lane) {
    const bf16* p = tile + lane * 16;
    v8bf lo = *(const v8bf*)(p);
    v8bf hi = *(const v8bf*)(p + 8);
    v16bf r;
#pragma unroll
    for (int j = 0; j < 8; ++j) { r[j] = lo[j]; r[j + 8] = hi[j]; }
    return r;
}

// ---------------------------------------------------------------------------
// Pack f32 weight matrix W[K x N] (row-major) into B-fragment tiles.
// Tile (tk, tn) -> 512 bf16 at P[(tn*TK + tk)*512]; within a tile,
// lane l elem e maps to W[tk*32 + ((l&16)?8:0) + e + (e&8)][tn*16 + (l&15)].
// ---------------------------------------------------------------------------
__global__ void pack_w_kernel(const float* __restrict__ W, bf16* __restrict__ P,
                              int K, int N) {
    const int  TK  = K >> 5;
    const long cnt = (long)TK * (N >> 4) * 32;
    const long idx = (long)blockIdx.x * 256 + threadIdx.x;
    if (idx >= cnt) return;
    const int  l  = (int)(idx & 31);
    const long ti = idx >> 5;
    const int  tk = (int)(ti % TK);
    const int  tn = (int)(ti / TK);
    const int  kb = tk * 32 + ((l & 16) ? 8 : 0);
    const int  n  = tn * 16 + (l & 15);
    bf16* dst = P + ti * 512 + l * 16;
    const float* src = W + (long)kb * N + n;
#pragma unroll
    for (int e = 0; e < 16; ++e) dst[e] = (bf16)src[(long)(e + (e & 8)) * N];
}

// ---------------------------------------------------------------------------
// Fused GLU GEMM: Yglu = GLU(relu(X @ W + b) * sc + bi)
//                 Y    = use_res ? (X + Yglu)*sqrt(0.5) : Yglu
// X,Y: [32768 x 512] bf16 ; Wpk: packed [512 x 1024] ; acc f32.
// Block tile: 128M x 64 GLU cols. Wave: 32M (2 A frags) x 32 GLU cols
// (gate col c and act col c+512 in same wave -> GLU is wave-local).
// ---------------------------------------------------------------------------
__global__ __launch_bounds__(256) void glu_gemm_kernel(
    const bf16* __restrict__ X, const bf16* __restrict__ Wpk,
    const float* __restrict__ b, const float* __restrict__ sc,
    const float* __restrict__ bi, bf16* __restrict__ Y, int use_res)
{
    constexpr int TK = FDIM / 32;   // 16 k-tiles
    __shared__ __align__(32) bf16 Alds[2][128 * LDS_A_STRIDE];

    const int tid  = threadIdx.x;
    const int lane = tid & 31;
    const int wave = tid >> 5;
    const int wm   = wave & 3;      // 32-row slab
    const int wn   = wave >> 2;     // 32 GLU cols
    const long m0  = (long)blockIdx.x * 128;
    const int  n0  = blockIdx.y * 64;           // GLU column base

    v8f acc[2][4] = {};             // [m half][t0,t1 gate ; t2,t3 act]

    // stage A tile tk into LDS buffer buf (128 rows x 32 k = 8KB, 2 async/thread)
    auto stage = [&](int tk, int buf) {
#pragma unroll
        for (int h = 0; h < 2; ++h) {
            const int ch = tid + h * 256;       // 512 chunks of 8 bf16
            const int r  = ch >> 2;
            const int c  = (ch & 3) * 8;
            async_copy16(&Alds[buf][r * LDS_A_STRIDE + c],
                         X + (m0 + r) * FDIM + tk * 32 + c);
        }
    };

    stage(0, 0);
    for (int tk = 0; tk < TK; ++tk) {
        const int cur = tk & 1;
        if (tk + 1 < TK) { stage(tk + 1, cur ^ 1); async_waitN<2>(); }
        else             { async_waitN<0>(); }
        __syncthreads();

        const bf16* Ab = &Alds[cur][0];
        const v16bf af0 = load_a_frag(Ab + (wm * 32 +  0) * LDS_A_STRIDE, lane);
        const v16bf af1 = load_a_frag(Ab + (wm * 32 + 16) * LDS_A_STRIDE, lane);
#pragma unroll
        for (int t = 0; t < 4; ++t) {
            const int gn = (t < 2) ? (n0 + wn * 32 + t * 16)
                                   : (512 + n0 + wn * 32 + (t - 2) * 16);
            const long ti = (long)(gn >> 4) * TK + tk;
            const v16bf bf_ = load_b_frag_g(Wpk + ti * 512, lane);
            acc[0][t] = __builtin_amdgcn_wmma_f32_16x16x32_bf16(
                false, af0, false, bf_, (short)0, acc[0][t], false, false);
            acc[1][t] = __builtin_amdgcn_wmma_f32_16x16x32_bf16(
                false, af1, false, bf_, (short)0, acc[1][t], false, false);
        }
        __syncthreads();
    }

    // epilogue: relu -> BN affine -> GLU -> (residual)
    const int nlane = lane & 15;
    const int rhalf = (lane < 16) ? 0 : 8;
#pragma unroll
    for (int mh = 0; mh < 2; ++mh) {
#pragma unroll
        for (int t = 0; t < 2; ++t) {
            const int cg = n0 + wn * 32 + t * 16 + nlane;   // gate col [0,512)
            const int ca = cg + 512;                        // act col
            const float bg = b[cg], sg = sc[cg], ig = bi[cg];
            const float ba = b[ca], sa = sc[ca], ia = bi[ca];
#pragma unroll
            for (int r = 0; r < 8; ++r) {
                const long row = m0 + wm * 32 + mh * 16 + rhalf + r;
                const float g  = fmaxf(acc[mh][t][r] + bg, 0.f) * sg + ig;
                const float a  = fmaxf(acc[mh][t + 2][r] + ba, 0.f) * sa + ia;
                const float glu = g * (1.f / (1.f + __expf(-a)));
                float out = glu;
                if (use_res) out = ((float)X[row * FDIM + cg] + glu) * SQRT_H;
                Y[row * FDIM + cg] = (bf16)out;
            }
        }
    }
}

// ---------------------------------------------------------------------------
// Generic MM kernel, block tile 128M x 128N, packed weights.
// MODE 0 (attention, X bf16): Y(f32) = (relu(X@W + b)*sc + bi) * prior
// MODE 1 (output,   X f32):   Y(f32) = X@W + b
// ---------------------------------------------------------------------------
template <int MODE, bool XBF>
__global__ __launch_bounds__(256) void mm_kernel(
    const void* __restrict__ Xv, int ldx, const bf16* __restrict__ Wpk,
    int K, const float* __restrict__ b, const float* __restrict__ sc,
    const float* __restrict__ bi, const float* __restrict__ prior, int ldp,
    float* __restrict__ Y, int ldy)
{
    __shared__ __align__(32) bf16 Alds[2][128 * LDS_A_STRIDE];

    const int tid  = threadIdx.x;
    const int lane = tid & 31;
    const int wave = tid >> 5;
    const int wm   = wave & 3;
    const int wn   = wave >> 2;
    const long m0  = (long)blockIdx.x * 128;
    const int  n0  = blockIdx.y * 128;
    const int  TK  = K >> 5;

    v8f acc[2][4] = {};

    auto stage_async = [&](int tk, int buf) {
#pragma unroll
        for (int h = 0; h < 2; ++h) {
            const int ch = tid + h * 256;
            const int r  = ch >> 2;
            const int c  = (ch & 3) * 8;
            async_copy16(&Alds[buf][r * LDS_A_STRIDE + c],
                         (const bf16*)Xv + (m0 + r) * (long)ldx + tk * 32 + c);
        }
    };
    auto stage_cvt = [&](int tk, int buf) {
#pragma unroll
        for (int h = 0; h < 2; ++h) {
            const int ch = tid + h * 256;
            const int r  = ch >> 2;
            const int c  = (ch & 3) * 8;
            bf16* dst = &Alds[buf][r * LDS_A_STRIDE + c];
            const float* src = (const float*)Xv + (m0 + r) * (long)ldx + tk * 32 + c;
#pragma unroll
            for (int j = 0; j < 8; ++j) dst[j] = (bf16)src[j];
        }
    };
    auto compute = [&](int tk, int cur) {
        const bf16* Ab = &Alds[cur][0];
        const v16bf af0 = load_a_frag(Ab + (wm * 32 +  0) * LDS_A_STRIDE, lane);
        const v16bf af1 = load_a_frag(Ab + (wm * 32 + 16) * LDS_A_STRIDE, lane);
#pragma unroll
        for (int t = 0; t < 4; ++t) {
            const int gn = n0 + wn * 64 + t * 16;
            const long ti = (long)(gn >> 4) * TK + tk;
            const v16bf bf_ = load_b_frag_g(Wpk + ti * 512, lane);
            acc[0][t] = __builtin_amdgcn_wmma_f32_16x16x32_bf16(
                false, af0, false, bf_, (short)0, acc[0][t], false, false);
            acc[1][t] = __builtin_amdgcn_wmma_f32_16x16x32_bf16(
                false, af1, false, bf_, (short)0, acc[1][t], false, false);
        }
    };

    if constexpr (XBF) {
        stage_async(0, 0);
        for (int tk = 0; tk < TK; ++tk) {
            const int cur = tk & 1;
            if (tk + 1 < TK) { stage_async(tk + 1, cur ^ 1); async_waitN<2>(); }
            else             { async_waitN<0>(); }
            __syncthreads();
            compute(tk, cur);
            __syncthreads();
        }
    } else {
        for (int tk = 0; tk < TK; ++tk) {
            stage_cvt(tk, 0);
            __syncthreads();
            compute(tk, 0);
            __syncthreads();
        }
    }

    const int nlane = lane & 15;
    const int rhalf = (lane < 16) ? 0 : 8;
#pragma unroll
    for (int mh = 0; mh < 2; ++mh) {
#pragma unroll
        for (int t = 0; t < 4; ++t) {
            const int c = n0 + wn * 64 + t * 16 + nlane;
            const float bc = b[c];
#pragma unroll
            for (int r = 0; r < 8; ++r) {
                const long row = m0 + wm * 32 + mh * 16 + rhalf + r;
                float v = acc[mh][t][r];
                if (MODE == 0) {
                    v = fmaxf(v + bc, 0.f) * sc[c] + bi[c];
                    v *= prior[row * (long)ldp + c];
                } else {
                    v += bc;
                }
                Y[row * (long)ldy + c] = v;
            }
        }
    }
}

// ---------------------------------------------------------------------------
// Sparsemax per row (512 cols). One 256-thread block per row.
// ---------------------------------------------------------------------------
__global__ __launch_bounds__(256) void sparsemax_kernel(
    const float* __restrict__ Z, float* __restrict__ prior,
    const bf16* __restrict__ normed, bf16* __restrict__ Xm,
    float* __restrict__ entropy)
{
    __shared__ float zorig[512];
    __shared__ float s[512];
    __shared__ float cs[512];
    __shared__ float tmp[512];
    __shared__ int   cnt;
    __shared__ float esum;

    const int  tid = threadIdx.x;
    const long row = blockIdx.x;
    const float* zr = Z + row * FDIM;

    for (int i = tid; i < 512; i += 256) { float v = zr[i]; zorig[i] = v; s[i] = v; }
    if (tid == 0) { cnt = 0; esum = 0.f; }
    __syncthreads();

    // bitonic sort, descending
    for (int ksz = 2; ksz <= 512; ksz <<= 1) {
        for (int j = ksz >> 1; j > 0; j >>= 1) {
            for (int i = tid; i < 512; i += 256) {
                const int ixj = i ^ j;
                if (ixj > i) {
                    const bool ddir = ((i & ksz) == 0);
                    const float a = s[i], c = s[ixj];
                    if (ddir ? (a < c) : (a > c)) { s[i] = c; s[ixj] = a; }
                }
            }
            __syncthreads();
        }
    }

    // inclusive prefix sum of sorted values
    for (int i = tid; i < 512; i += 256) cs[i] = s[i];
    __syncthreads();
    for (int off = 1; off < 512; off <<= 1) {
        for (int i = tid; i < 512; i += 256)
            tmp[i] = (i >= off) ? (cs[i - off] + cs[i]) : cs[i];
        __syncthreads();
        for (int i = tid; i < 512; i += 256) cs[i] = tmp[i];
        __syncthreads();
    }

    // support size
    for (int i = tid; i < 512; i += 256) {
        const float k = (float)(i + 1);
        if (1.0f + k * s[i] > cs[i]) atomicAdd(&cnt, 1);
    }
    __syncthreads();
    const int   kz  = (cnt > 0) ? cnt : 1;
    const float tau = (cs[kz - 1] - 1.0f) / (float)kz;

    float eloc = 0.f;
    for (int i = tid; i < 512; i += 256) {
        const float m  = fmaxf(0.f, zorig[i] - tau);
        const long idx = row * FDIM + i;
        prior[idx] = prior[idx] * (1.5f - m);            // GAMMA = 1.5
        Xm[idx]    = (bf16)((float)normed[idx] * m);
        eloc      += -m * __logf(m + 1e-5f);
    }
    atomicAdd(&esum, eloc);
    __syncthreads();
    if (tid == 0)
        atomicAdd(entropy, esum / ((float)NSTEPS * (float)BATCH));
}

// ---------------------------------------------------------------------------
// Small elementwise kernels
// ---------------------------------------------------------------------------
__global__ void bn_in_kernel(const float* __restrict__ data, const float* __restrict__ sc,
                             const float* __restrict__ bi, bf16* __restrict__ out) {
    const long i = (long)blockIdx.x * 256 + threadIdx.x;   // over B*512
    const int  c = (int)(i & 511);
    out[i] = (bf16)(data[i] * sc[c] + bi[c]);
}

__global__ void init_kernel(float* __restrict__ prior, float* __restrict__ decision,
                            float* __restrict__ entropy) {
    const long i = (long)blockIdx.x * 256 + threadIdx.x;   // over B*512
    prior[i] = 1.0f;
    if (i < BATCH * ADIM) decision[i] = 0.0f;
    if (i == 0) *entropy = 0.0f;
}

__global__ void dec_update_kernel(const bf16* __restrict__ x, float* __restrict__ dec) {
    const long i   = (long)blockIdx.x * 256 + threadIdx.x; // over B*256
    const long row = i >> 8;
    const int  c   = (int)(i & 255);
    dec[i] += fmaxf((float)x[row * FDIM + c], 0.f);
}

// ---------------------------------------------------------------------------
// Host orchestration
// ---------------------------------------------------------------------------
extern "C" void kernel_launch(void* const* d_in, const int* in_sizes, int n_in,
                              void* d_out, int out_size, void* d_ws, size_t ws_size,
                              hipStream_t stream) {
    (void)in_sizes; (void)n_in; (void)out_size; (void)ws_size;

    const float* data      = (const float*)d_in[0];
    const float* in_sc     = (const float*)d_in[1];
    const float* in_bi     = (const float*)d_in[2];
    const float* shared_W0 = (const float*)d_in[3];
    const float* shared_b0 = (const float*)d_in[4];
    const float* shared_W1 = (const float*)d_in[5];
    const float* shared_b1 = (const float*)d_in[6];
    const float* shared_sc = (const float*)d_in[7];   // [2,1024]
    const float* shared_bi = (const float*)d_in[8];
    const float* feat_W    = (const float*)d_in[9];   // [6,2,512,1024]
    const float* feat_b    = (const float*)d_in[10];  // [6,2,1024]
    const float* feat_sc   = (const float*)d_in[11];
    const float* feat_bi   = (const float*)d_in[12];
    const float* attn_W    = (const float*)d_in[13];  // [5,256,512]
    const float* attn_b    = (const float*)d_in[14];  // [5,512]
    const float* attn_sc   = (const float*)d_in[15];
    const float* attn_bi   = (const float*)d_in[16];
    const float* out_W     = (const float*)d_in[17];  // [256,128]
    const float* out_b     = (const float*)d_in[18];

    float* out_mat = (float*)d_out;                   // [32768 x 128]
    float* entropy = out_mat + BATCH * ODIM;          // scalar

    // workspace carve-up
    char* ws = (char*)d_ws;
    size_t off = 0;
    auto carve = [&](size_t bytes) { void* p = ws + off; off += (bytes + 255) & ~255ULL; return p; };
    bf16* sW0p  = (bf16*)carve(512L * 1024 * 2);
    bf16* sW1p  = (bf16*)carve(512L * 1024 * 2);
    bf16* fWp   = (bf16*)carve(6L * 2 * 512 * 1024 * 2);
    bf16* aWp   = (bf16*)carve(5L * 256 * 512 * 2);
    bf16* oWp   = (bf16*)carve(256L * 128 * 2);
    bf16*  normed = (bf16*)carve(BATCH * FDIM * 2);
    bf16*  xA     = (bf16*)carve(BATCH * FDIM * 2);
    bf16*  xB     = (bf16*)carve(BATCH * FDIM * 2);
    bf16*  xm     = (bf16*)carve(BATCH * FDIM * 2);
    float* prior  = (float*)carve(BATCH * FDIM * 4);
    float* zbuf   = (float*)carve(BATCH * FDIM * 4);
    float* dec    = (float*)carve(BATCH * ADIM * 4);

    // weight packing into B-fragment layout (L2-resident, cheap)
    auto pack = [&](const float* W, bf16* P, int K, int N) {
        const long cnt = (long)(K >> 5) * (N >> 4) * 32;
        pack_w_kernel<<<(cnt + 255) / 256, 256, 0, stream>>>(W, P, K, N);
    };
    pack(shared_W0, sW0p, FDIM, 1024);
    pack(shared_W1, sW1p, FDIM, 1024);
    for (int s = 0; s < 12; ++s)
        pack(feat_W + (long)s * FDIM * 1024, fWp + (long)s * FDIM * 1024, FDIM, 1024);
    for (int s = 0; s < NSTEPS; ++s)
        pack(attn_W + (long)s * ADIM * FDIM, aWp + (long)s * ADIM * FDIM, ADIM, FDIM);
    pack(out_W, oWp, ADIM, ODIM);

    // input BN + init
    bn_in_kernel<<<BATCH * FDIM / 256, 256, 0, stream>>>(data, in_sc, in_bi, normed);
    init_kernel<<<BATCH * FDIM / 256, 256, 0, stream>>>(prior, dec, entropy);

    const dim3 glu_grid(BATCH / 128, FDIM / 64);       // (256, 8)
    const dim3 attn_grid(BATCH / 128, FDIM / 128);     // (256, 4)
    const dim3 out_grid(BATCH / 128, ODIM / 128);      // (256, 1)

    // feature transformer: Xin -> xB (ping-pong xA/xB)
    auto feat_transform = [&](const bf16* Xin, int step) {
        glu_gemm_kernel<<<glu_grid, 256, 0, stream>>>(
            Xin, sW0p, shared_b0, shared_sc + 0 * 1024, shared_bi + 0 * 1024, xA, 0);
        glu_gemm_kernel<<<glu_grid, 256, 0, stream>>>(
            xA, sW1p, shared_b1, shared_sc + 1 * 1024, shared_bi + 1 * 1024, xB, 1);
        const long wstep = (long)step * 2 * FDIM * 1024;
        const long pstep = (long)step * 2 * 1024;
        glu_gemm_kernel<<<glu_grid, 256, 0, stream>>>(
            xB, fWp + wstep, feat_b + pstep, feat_sc + pstep, feat_bi + pstep, xA, 1);
        glu_gemm_kernel<<<glu_grid, 256, 0, stream>>>(
            xA, fWp + wstep + (long)FDIM * 1024, feat_b + pstep + 1024,
            feat_sc + pstep + 1024, feat_bi + pstep + 1024, xB, 1);
    };

    // step 0 transformer on normed input; a_i = xB[:, 256:512]
    feat_transform(normed, 0);

    for (int i = 0; i < NSTEPS; ++i) {
        // z = (relu(a @ attn_W[i] + b)*sc + bi) * prior
        mm_kernel<0, true><<<attn_grid, 256, 0, stream>>>(
            xB + ADIM, FDIM, aWp + (long)i * ADIM * FDIM, ADIM,
            attn_b + (long)i * FDIM, attn_sc + (long)i * FDIM,
            attn_bi + (long)i * FDIM, prior, FDIM, zbuf, FDIM);

        // mask = sparsemax(z); prior *= (gamma - mask); xm = normed*mask; entropy +=
        sparsemax_kernel<<<BATCH, 256, 0, stream>>>(zbuf, prior, normed, xm, entropy);

        // next transformer
        feat_transform(xm, i + 1);

        // decision += relu(xB[:, :256])
        dec_update_kernel<<<BATCH * ADIM / 256, 256, 0, stream>>>(xB, dec);
    }

    // out = decision @ out_W + out_b
    mm_kernel<1, false><<<out_grid, 256, 0, stream>>>(
        dec, ADIM, oWp, ADIM, out_b,
        nullptr, nullptr, nullptr, 0, out_mat, ODIM);
}
